// SignalSligtlyOptimizedFlipFlopLayer_6906307412602
// MI455X (gfx1250) — matrix operand order repaired and checked by reading, not compiled, not run on hardware
//
#include <hip/hip_runtime.h>

// ---------------------------------------------------------------------------
// Flip-flop RNN, fused persistent-state kernel for gfx1250 (MI455X).
// B=512, I=512, H=1024, O=512, T=128.
// - 32 workgroups x 1024 threads; each WG owns 16 batch rows for all T steps
//   (rows are independent through the recurrence).
// - h: fp32 fragments in registers (native D-matrix layout), bf16 copy
//   double-buffered in 64KB LDS for the WMMA A operands; 1 barrier/step.
// - Weights bf16, streamed from L2 (5MB << 192MB), V_WMMA_F32_16X16X32_BF16.
// - jx/kx (+biases) fragments stashed wave-private in global ws in fragment
//   layout; added AFTER the WMMA k-loop so they never occupy registers
//   during the loop (kills the spills seen in round 1).
// - Output stream (128MB) written with non-temporal stores.
// ---------------------------------------------------------------------------

constexpr int Bdim = 512;
constexpr int Idim = 512;
constexpr int Hdim = 1024;
constexpr int Odim = 512;

typedef __bf16 v16bf __attribute__((ext_vector_type(16)));
typedef __bf16 v8bf  __attribute__((ext_vector_type(8)));
typedef float  v8f   __attribute__((ext_vector_type(8)));

__device__ __forceinline__ unsigned short f2bf(float f) {
  unsigned int u = __float_as_uint(f);
  u += 0x7FFFu + ((u >> 16) & 1u);       // round-to-nearest-even
  return (unsigned short)(u >> 16);
}

__device__ __forceinline__ v8bf ld8(const unsigned short* p) {
  return *reinterpret_cast<const v8bf*>(p);   // one 16-byte load
}

__device__ __forceinline__ v16bf mk16(v8bf lo, v8bf hi) {
  return __builtin_shufflevector(lo, hi, 0,1,2,3,4,5,6,7,8,9,10,11,12,13,14,15);
}

__device__ __forceinline__ v8f wmma_bf16(v16bf a, v16bf b, v8f c) {
  // D = A(16x32 bf16) * B(32x16 bf16) + C(16x16 f32)
  return __builtin_amdgcn_wmma_f32_16x16x32_bf16(false, a, false, b,
                                                 (short)0, c, false, false);
}

__device__ __forceinline__ float sigmoidf(float x) {
  return 1.0f / (1.0f + __expf(-x));
}

// --------------------------- fp32 -> bf16 conversion -----------------------
__global__ void cvt_f32_bf16(const float* __restrict__ src,
                             unsigned short* __restrict__ dst, int n) {
  int i = blockIdx.x * blockDim.x + threadIdx.x;
  int stride = gridDim.x * blockDim.x;
  for (; i < n; i += stride) dst[i] = f2bf(src[i]);
}

// ----------------------------- fused recurrence ----------------------------
__global__ __launch_bounds__(1024, 1)
void flipflop_fused(const float* __restrict__ h0,
                    const float* __restrict__ bjx, const float* __restrict__ bjh,
                    const float* __restrict__ bkx, const float* __restrict__ bkh,
                    const float* __restrict__ bo,
                    const unsigned short* __restrict__ xb,
                    const unsigned short* __restrict__ Wjxb,
                    const unsigned short* __restrict__ Wkxb,
                    const unsigned short* __restrict__ Wjhb,
                    const unsigned short* __restrict__ Wkhb,
                    const unsigned short* __restrict__ Wob,
                    float* __restrict__ jkstash,
                    const int* __restrict__ seqlen,
                    float* __restrict__ out) {
  // double-buffered bf16 h tile: 2 * 16 * 1024 * 2B = 64 KB LDS
  __shared__ unsigned short hbuf[2][16 * Hdim];

  const int T     = seqlen[0];
  const int lane  = threadIdx.x & 31;
  const int wave  = threadIdx.x >> 5;        // 0..31
  const int n     = lane & 15;               // column within 16-wide tile
  const int half  = lane >> 4;               // 0: lanes 0-15, 1: lanes 16-31
  const int m0    = half * 8;                // D-layout row base for this half
  const int bBase = blockIdx.x << 4;         // 16 batch rows per workgroup
  const int o0    = wave * 16;               // this wave's output column tile

  // Wave-private stash of the 4 accumulator-init fragments (J0,J1,K0,K1),
  // fragment-major: 32 lanes x 8 floats = 1KB contiguous per fragment.
  float* stash = jkstash + ((size_t)(blockIdx.x * 32 + wave) * 4) * 256
                         + (size_t)lane * 8;

  v8f hreg[2];   // persistent fp32 h fragments (only 16 VGPRs)

  // ---- phase 0: jx + bjx + bjh, kx + bkx + bkh, and initial h --------------
#pragma unroll
  for (int i = 0; i < 2; ++i) {
    const int c = wave * 32 + i * 16;        // H-column tile base
    const unsigned short* xrow  = xb   + (size_t)(bBase + n) * Idim;
    const unsigned short* wjrow = Wjxb + (size_t)(c + n) * Idim;
    const unsigned short* wkrow = Wkxb + (size_t)(c + n) * Idim;
    v8f aj = {0.f,0.f,0.f,0.f,0.f,0.f,0.f,0.f};
    v8f ak = {0.f,0.f,0.f,0.f,0.f,0.f,0.f,0.f};
    for (int k = 0; k < Idim; k += 32) {
      v16bf a = mk16(ld8(xrow + k + half * 16), ld8(xrow + k + half * 16 + 8));
      v16bf bj = mk16(ld8(wjrow + k + half * 16), ld8(wjrow + k + half * 16 + 8));
      aj = wmma_bf16(a, bj, aj);
      v16bf bk = mk16(ld8(wkrow + k + half * 16), ld8(wkrow + k + half * 16 + 8));
      ak = wmma_bf16(a, bk, ak);
    }
    const float addj = bjx[c + n] + bjh[c + n];
    const float addk = bkx[c + n] + bkh[c + n];
    v8f h;
#pragma unroll
    for (int r = 0; r < 8; ++r) {
      aj[r] += addj;
      ak[r] += addk;
      h[r] = h0[(size_t)(bBase + m0 + r) * Hdim + c + n];
      hbuf[0][(m0 + r) * Hdim + c + n] = f2bf(h[r]);
    }
    hreg[i] = h;
    // wave-private stash (same-wave store->load is kept in order by HW)
    *reinterpret_cast<v8f*>(stash + (size_t)(i)     * 256) = aj;
    *reinterpret_cast<v8f*>(stash + (size_t)(i + 2) * 256) = ak;
  }
  const float bov = bo[o0 + n];

  // Per-lane weight-row pointers (fixed for the whole time loop)
  const unsigned short* wjr0 = Wjhb + (size_t)(wave * 32 + n) * Hdim;
  const unsigned short* wjr1 = Wjhb + (size_t)(wave * 32 + 16 + n) * Hdim;
  const unsigned short* wkr0 = Wkhb + (size_t)(wave * 32 + n) * Hdim;
  const unsigned short* wkr1 = Wkhb + (size_t)(wave * 32 + 16 + n) * Hdim;
  const unsigned short* wor  = Wob  + (size_t)(o0 + n) * Hdim;

  __syncthreads();

  // ---- time loop -----------------------------------------------------------
  for (int t = 0; t < T; ++t) {
    const int cur = t & 1, nxt = cur ^ 1;

    // phase 1: acc = h @ W^T   (jx/kx init added after the loop)
    v8f aJ0 = {0.f,0.f,0.f,0.f,0.f,0.f,0.f,0.f};
    v8f aJ1 = aJ0, aK0 = aJ0, aK1 = aJ0;
    const unsigned short* hrow = &hbuf[cur][n * Hdim];
#pragma unroll 2
    for (int k = 0; k < Hdim; k += 32) {
      v16bf a = mk16(ld8(hrow + k + half * 16), ld8(hrow + k + half * 16 + 8));
      v16bf b0 = mk16(ld8(wjr0 + k + half * 16), ld8(wjr0 + k + half * 16 + 8));
      aJ0 = wmma_bf16(a, b0, aJ0);
      v16bf b1 = mk16(ld8(wjr1 + k + half * 16), ld8(wjr1 + k + half * 16 + 8));
      aJ1 = wmma_bf16(a, b1, aJ1);
      v16bf b2 = mk16(ld8(wkr0 + k + half * 16), ld8(wkr0 + k + half * 16 + 8));
      aK0 = wmma_bf16(a, b2, aK0);
      v16bf b3 = mk16(ld8(wkr1 + k + half * 16), ld8(wkr1 + k + half * 16 + 8));
      aK1 = wmma_bf16(a, b3, aK1);
    }
    // add the stashed jx/kx(+bias) fragments now (fp32 add is associative
    // enough; keeps 32 VGPRs free during the whole k-loop)
    aJ0 += *reinterpret_cast<const v8f*>(stash + 0 * 256);
    aJ1 += *reinterpret_cast<const v8f*>(stash + 1 * 256);
    aK0 += *reinterpret_cast<const v8f*>(stash + 2 * 256);
    aK1 += *reinterpret_cast<const v8f*>(stash + 3 * 256);

    // phase 2: gate + update, write new h (bf16) into the other buffer
#pragma unroll
    for (int i = 0; i < 2; ++i) {
      const int c = wave * 32 + i * 16;
      const v8f aJ = i ? aJ1 : aJ0;
      const v8f aK = i ? aK1 : aK0;
      v8f h = hreg[i];
#pragma unroll
      for (int r = 0; r < 8; ++r) {
        float j = sigmoidf(aJ[r]);
        float q = sigmoidf(aK[r]);
        float hn = j + h[r] * (1.0f - j - q);   // == j*(1-h) + (1-q)*h
        h[r] = hn;
        hbuf[nxt][(m0 + r) * Hdim + c + n] = f2bf(hn);
      }
      hreg[i] = h;
    }
    __syncthreads();   // one barrier per step (double-buffered h)

    // phase 3: out[:, t, :] = new_h @ Wo^T + bo
    v8f acc = { bov, bov, bov, bov, bov, bov, bov, bov };
    const unsigned short* hrow2 = &hbuf[nxt][n * Hdim];
    for (int k = 0; k < Hdim; k += 32) {
      v16bf a = mk16(ld8(hrow2 + k + half * 16), ld8(hrow2 + k + half * 16 + 8));
      v16bf b = mk16(ld8(wor + k + half * 16), ld8(wor + k + half * 16 + 8));
      acc = wmma_bf16(a, b, acc);
    }
    float* op = out + (size_t)(bBase + m0) * ((size_t)T * Odim)
                    + (size_t)t * Odim + o0 + n;
#pragma unroll
    for (int r = 0; r < 8; ++r)
      __builtin_nontemporal_store(acc[r], op + (size_t)r * (size_t)T * Odim);
  }

  // ---- h_final (second tuple element), fp32, [B,H] ------------------------
  float* hfin = out + (size_t)Bdim * (size_t)T * Odim;
#pragma unroll
  for (int i = 0; i < 2; ++i) {
    const int c = wave * 32 + i * 16;
#pragma unroll
    for (int r = 0; r < 8; ++r)
      __builtin_nontemporal_store(hreg[i][r],
          hfin + (size_t)(bBase + m0 + r) * Hdim + c + n);
  }
}

// ------------------------------- launcher ----------------------------------
extern "C" void kernel_launch(void* const* d_in, const int* in_sizes, int n_in,
                              void* d_out, int out_size, void* d_ws, size_t ws_size,
                              hipStream_t stream) {
  (void)in_sizes; (void)n_in; (void)out_size; (void)ws_size;

  const float* x    = (const float*)d_in[0];
  const float* h0   = (const float*)d_in[1];
  const float* Wjx  = (const float*)d_in[2];
  const float* bjx  = (const float*)d_in[3];
  const float* Wjh  = (const float*)d_in[4];
  const float* bjh  = (const float*)d_in[5];
  const float* Wkx  = (const float*)d_in[6];
  const float* bkx  = (const float*)d_in[7];
  const float* Wkh  = (const float*)d_in[8];
  const float* bkh  = (const float*)d_in[9];
  const float* Wo   = (const float*)d_in[10];
  const float* bo   = (const float*)d_in[11];
  const int*   seqT = (const int*)d_in[12];
  float* out = (float*)d_out;

  // Workspace layout: bf16 tensors (7.5 MB) then fp32 jx/kx stash (4 MB)
  unsigned short* ws = (unsigned short*)d_ws;
  unsigned short* xb   = ws;                          size_t off = (size_t)Bdim * Idim;
  unsigned short* Wjxb = ws + off;                    off += (size_t)Hdim * Idim;
  unsigned short* Wkxb = ws + off;                    off += (size_t)Hdim * Idim;
  unsigned short* Wjhb = ws + off;                    off += (size_t)Hdim * Hdim;
  unsigned short* Wkhb = ws + off;                    off += (size_t)Hdim * Hdim;
  unsigned short* Wob  = ws + off;                    off += (size_t)Odim * Hdim;
  float* jkstash = (float*)(ws + off);                // 32*32*4*256 floats = 4MB

  auto cvt = [&](const float* s, unsigned short* d, int cnt) {
    int blocks = (cnt + 255) / 256;
    cvt_f32_bf16<<<dim3(blocks), dim3(256), 0, stream>>>(s, d, cnt);
  };
  cvt(x,   xb,   Bdim * Idim);
  cvt(Wjx, Wjxb, Hdim * Idim);
  cvt(Wkx, Wkxb, Hdim * Idim);
  cvt(Wjh, Wjhb, Hdim * Hdim);
  cvt(Wkh, Wkhb, Hdim * Hdim);
  cvt(Wo,  Wob,  Odim * Hdim);

  flipflop_fused<<<dim3(Bdim / 16), dim3(1024), 0, stream>>>(
      h0, bjx, bjh, bkx, bkh, bo,
      xb, Wjxb, Wkxb, Wjhb, Wkhb, Wob, jkstash, seqT, out);
}
